// GQABlock_89223650607442
// MI455X (gfx1250) — matrix operand order repaired and checked by reading, not compiled, not run on hardware
//
#include <hip/hip_runtime.h>
#include <math.h>

typedef __attribute__((ext_vector_type(16))) _Float16 v16h;
typedef __attribute__((ext_vector_type(8)))  float    v8f;

#define B_    2
#define T_    2048
#define D_    1536
#define NH_   12
#define NKV_  4
#define HD_   128
#define NREP_ 3
#define FI_   4096
#define CHUNK_ 256
#define KW_   5
#define TTT_LR_ 0.01f
#define EPS_  1e-6f

// ---------------------------------------------------------------- WMMA helpers
__device__ __forceinline__ v8f wmma_f16(v16h a, v16h b, v8f c) {
  return __builtin_amdgcn_wmma_f32_16x16x32_f16(false, a, false, b, (short)0, c,
                                                false, false);
}

// A-fragment (16x32 f16, row-major tile in LDS): lanes 0-15 rows M, K halves by lane>>4
__device__ __forceinline__ v16h frag_a(const _Float16* p_, int ld, int m0, int k0, int lane) {
  const int r = lane & 15, kh = (lane >> 4) & 1;
  const _Float16* p = p_ + (m0 + r) * ld + k0;
  v16h o;
#pragma unroll
  for (int i = 0; i < 8; ++i) { o[i] = p[kh * 8 + i]; o[8 + i] = p[16 + kh * 8 + i]; }
  return o;
}

// B-fragment (32x16 f16) from weight-layout [N][K] tile in LDS: lane = column N,
// K halves by lane>>4 (same order as frag_a)
__device__ __forceinline__ v16h frag_b(const _Float16* p_, int ld, int n0, int k0, int lane) {
  const int n = lane & 15, kh = (lane >> 4) & 1;
  const _Float16* p = p_ + (n0 + n) * ld + k0;
  v16h o;
#pragma unroll
  for (int j = 0; j < 8; ++j) { o[j] = p[kh * 8 + j]; o[8 + j] = p[16 + kh * 8 + j]; }
  return o;
}

// ---------------------------------------------------------------- GEMM C = alpha*A.B^T (+C)
// A[M,K], B[N,K] row-major fp32 in global, f16 in WMMA, fp32 accumulate.
#define BM 128
#define BN 128
#define BK 32
#define LDT (BK + 8)

__global__ __launch_bounds__(256) void gemm_nt(
    const float* __restrict__ A, const float* __restrict__ Bm, float* __restrict__ C,
    int M, int N, int K, long sA, long sB, long sC, float alpha, int accum) {
  __shared__ _Float16 As[BM * LDT];
  __shared__ _Float16 Bs[BN * LDT];
  A  += (long)blockIdx.z * sA;
  Bm += (long)blockIdx.z * sB;
  C  += (long)blockIdx.z * sC;
  const int tid = threadIdx.x, lane = tid & 31, w = tid >> 5;
  const int wm = (w >> 2) * 64, wn = (w & 3) * 32;
  const int m0 = blockIdx.y * BM, n0 = blockIdx.x * BN;
  const int lr = tid >> 1, lc = (tid & 1) * 16;
  v8f acc[4][2];
#pragma unroll
  for (int i = 0; i < 4; ++i)
#pragma unroll
    for (int j = 0; j < 2; ++j) acc[i][j] = {};

  for (int k0 = 0; k0 < K; k0 += BK) {
    const float* ag = A  + (long)(m0 + lr) * K + k0 + lc;
    const float* bg = Bm + (long)(n0 + lr) * K + k0 + lc;
    if (k0 + BK < K) {                       // emits global_prefetch_b8
      __builtin_prefetch(ag + BK, 0, 1);
      __builtin_prefetch(bg + BK, 0, 1);
    }
#pragma unroll
    for (int j = 0; j < 16; ++j) {
      As[lr * LDT + lc + j] = (_Float16)ag[j];
      Bs[lr * LDT + lc + j] = (_Float16)bg[j];
    }
    __syncthreads();
    v16h a4[4], b2[2];
#pragma unroll
    for (int mi = 0; mi < 4; ++mi) a4[mi] = frag_a(As, LDT, wm + mi * 16, 0, lane);
#pragma unroll
    for (int ni = 0; ni < 2; ++ni) b2[ni] = frag_b(Bs, LDT, wn + ni * 16, 0, lane);
#pragma unroll
    for (int mi = 0; mi < 4; ++mi)
#pragma unroll
      for (int ni = 0; ni < 2; ++ni) acc[mi][ni] = wmma_f16(a4[mi], b2[ni], acc[mi][ni]);
    __syncthreads();
  }
  const int cr = 8 * (lane >> 4), cc = lane & 15;
  if (accum) {           // uniform branch hoisted: batched RMW stores
#pragma unroll
    for (int mi = 0; mi < 4; ++mi)
#pragma unroll
      for (int ni = 0; ni < 2; ++ni) {
        const int col = n0 + wn + ni * 16 + cc;
#pragma unroll
        for (int v = 0; v < 8; ++v) {
          const int row = m0 + wm + mi * 16 + cr + v;
          const long idx = (long)row * N + col;
          C[idx] += alpha * acc[mi][ni][v];
        }
      }
  } else {
#pragma unroll
    for (int mi = 0; mi < 4; ++mi)
#pragma unroll
      for (int ni = 0; ni < 2; ++ni) {
        const int col = n0 + wn + ni * 16 + cc;
#pragma unroll
        for (int v = 0; v < 8; ++v) {
          const int row = m0 + wm + mi * 16 + cr + v;
          C[(long)row * N + col] = alpha * acc[mi][ni][v];
        }
      }
  }
}

// ------------------------------------------------- GEMM C += alpha * A^T.B
// A[K,M], B[K,N] row-major fp32 (transposed gathers into LDS), C[M,N] read-modify-write.
__global__ __launch_bounds__(256) void gemm_tn_acc(
    const float* __restrict__ A, const float* __restrict__ Bm, float* __restrict__ C,
    int M, int N, int K, long sA, long sB, long sC, float alpha) {
  __shared__ _Float16 As[BM * LDT];
  __shared__ _Float16 Bs[BN * LDT];
  A  += (long)blockIdx.z * sA;
  Bm += (long)blockIdx.z * sB;
  C  += (long)blockIdx.z * sC;
  const int tid = threadIdx.x, lane = tid & 31, w = tid >> 5;
  const int wm = (w >> 2) * 64, wn = (w & 3) * 32;
  const int m0 = blockIdx.y * BM, n0 = blockIdx.x * BN;
  const int lr = tid >> 1, lc = (tid & 1) * 16;
  v8f acc[4][2];
#pragma unroll
  for (int i = 0; i < 4; ++i)
#pragma unroll
    for (int j = 0; j < 2; ++j) acc[i][j] = {};

  for (int k0 = 0; k0 < K; k0 += BK) {
#pragma unroll
    for (int j = 0; j < 16; ++j) {
      As[lr * LDT + lc + j] = (_Float16)A[(long)(k0 + lc + j) * M + m0 + lr];
      Bs[lr * LDT + lc + j] = (_Float16)Bm[(long)(k0 + lc + j) * N + n0 + lr];
    }
    __syncthreads();
    v16h a4[4], b2[2];
#pragma unroll
    for (int mi = 0; mi < 4; ++mi) a4[mi] = frag_a(As, LDT, wm + mi * 16, 0, lane);
#pragma unroll
    for (int ni = 0; ni < 2; ++ni) b2[ni] = frag_b(Bs, LDT, wn + ni * 16, 0, lane);
#pragma unroll
    for (int mi = 0; mi < 4; ++mi)
#pragma unroll
      for (int ni = 0; ni < 2; ++ni) acc[mi][ni] = wmma_f16(a4[mi], b2[ni], acc[mi][ni]);
    __syncthreads();
  }
  const int cr = 8 * (lane >> 4), cc = lane & 15;
#pragma unroll
  for (int mi = 0; mi < 4; ++mi)
#pragma unroll
    for (int ni = 0; ni < 2; ++ni) {
      const int col = n0 + wn + ni * 16 + cc;
#pragma unroll
      for (int v = 0; v < 8; ++v) {
        const int row = m0 + wm + mi * 16 + cr + v;
        C[(long)row * N + col] += alpha * acc[mi][ni][v];
      }
    }
}

// ---------------------------------------------------------------- flash attention (causal GQA)
// Per block: one (b, h, 64-query tile). 64-key blocks, online softmax, WMMA QK^T and PV.
__global__ __launch_bounds__(256) void attn_fa(
    const float* __restrict__ q, const float* __restrict__ k,
    const float* __restrict__ v, float* __restrict__ y) {
  const int qt = blockIdx.x, h = blockIdx.y, b = blockIdx.z;
  const int kvh = h / NREP_;
  const int tid = threadIdx.x, lane = tid & 31, w = tid >> 5;
  const int mt = w & 3, cg = w >> 2;          // wave tile: rows mt*16, col group cg
  __shared__ _Float16 Qs[64 * (HD_ + 8)];
  __shared__ _Float16 Ks[64 * (HD_ + 8)];
  __shared__ _Float16 Vs[HD_ * (64 + 8)];     // [d][key]  (weight layout for PV)
  __shared__ _Float16 Ps[64 * (64 + 8)];
  __shared__ float Sb[64 * 65];
  __shared__ float mrow[64], lrow[64], arow[64];
  const int qbase = qt * 64;
  const float scale = 0.08838834764831845f;   // 1/sqrt(128)
  {
    const int r = tid >> 2, c0 = (tid & 3) * 32;
    const float* src = q + (((long)(b * T_ + qbase + r)) * NH_ + h) * HD_ + c0;
#pragma unroll
    for (int j = 0; j < 32; ++j) Qs[r * (HD_ + 8) + c0 + j] = (_Float16)(src[j] * scale);
  }
  if (tid < 64) { mrow[tid] = -1e30f; lrow[tid] = 0.f; }
  v8f accO[4];
#pragma unroll
  for (int i = 0; i < 4; ++i) accO[i] = {};
  __syncthreads();

  for (int kb = 0; kb <= qt; ++kb) {
    {
      const int r = tid >> 2, c0 = (tid & 3) * 32;
      const float* ks = k + (((long)(b * T_ + kb * 64 + r)) * NKV_ + kvh) * HD_ + c0;
#pragma unroll
      for (int j = 0; j < 32; ++j) Ks[r * (HD_ + 8) + c0 + j] = (_Float16)ks[j];
      const int kv = tid & 63, d0 = (tid >> 6) * 32;
      const float* vs = v + (((long)(b * T_ + kb * 64 + kv)) * NKV_ + kvh) * HD_ + d0;
#pragma unroll
      for (int j = 0; j < 32; ++j) Vs[(d0 + j) * (64 + 8) + kv] = (_Float16)vs[j];
    }
    __syncthreads();
    // S = Q.K^T   (each wave: 16x32 of the 64x64 score tile)
    v8f sc[2]; sc[0] = {}; sc[1] = {};
#pragma unroll
    for (int kc = 0; kc < 4; ++kc) {
      v16h a = frag_a(Qs, HD_ + 8, mt * 16, kc * 32, lane);
#pragma unroll
      for (int ni = 0; ni < 2; ++ni) {
        v16h bb = frag_b(Ks, HD_ + 8, cg * 32 + ni * 16, kc * 32, lane);
        sc[ni] = wmma_f16(a, bb, sc[ni]);
      }
    }
    {
      const int cr = 8 * (lane >> 4), cc = lane & 15;
#pragma unroll
      for (int ni = 0; ni < 2; ++ni) {
        const int c = cg * 32 + ni * 16 + cc, gc = kb * 64 + c;
#pragma unroll
        for (int vv = 0; vv < 8; ++vv) {
          const int r = mt * 16 + cr + vv, gr = qbase + r;
          Sb[r * 65 + c] = (gc <= gr) ? sc[ni][vv] : -1e30f;   // causal mask
        }
      }
    }
    __syncthreads();
    if (tid < 64) {                       // online softmax, one thread per row
      const int r = tid;
      float mx = mrow[r];
      for (int c = 0; c < 64; ++c) mx = fmaxf(mx, Sb[r * 65 + c]);
      const float al = __expf(mrow[r] - mx);
      float s = 0.f;
      for (int c = 0; c < 64; ++c) {
        const float p = __expf(Sb[r * 65 + c] - mx);
        s += p;
        Ps[r * (64 + 8) + c] = (_Float16)p;
      }
      mrow[r] = mx; lrow[r] = lrow[r] * al + s; arow[r] = al;
    }
    __syncthreads();
    {
      const int cr = 8 * (lane >> 4);
#pragma unroll
      for (int oi = 0; oi < 4; ++oi)
#pragma unroll
        for (int vv = 0; vv < 8; ++vv) accO[oi][vv] *= arow[mt * 16 + cr + vv];
    }
#pragma unroll
    for (int kc = 0; kc < 2; ++kc) {      // O += P.V
      v16h a = frag_a(Ps, 64 + 8, mt * 16, kc * 32, lane);
#pragma unroll
      for (int oi = 0; oi < 4; ++oi) {
        v16h bb = frag_b(Vs, 64 + 8, cg * 64 + oi * 16, kc * 32, lane);
        accO[oi] = wmma_f16(a, bb, accO[oi]);
      }
    }
    __syncthreads();
  }
  {
    const int cr = 8 * (lane >> 4), cc = lane & 15;
#pragma unroll
    for (int oi = 0; oi < 4; ++oi) {
      const int d = cg * 64 + oi * 16 + cc;
#pragma unroll
      for (int vv = 0; vv < 8; ++vv) {
        const int r = mt * 16 + cr + vv;
        y[(long)(b * T_ + qbase + r) * (NH_ * HD_) + h * HD_ + d] = accO[oi][vv] / lrow[r];
      }
    }
  }
}

// ---------------------------------------------------------------- elementwise kernels
__global__ __launch_bounds__(256) void rmsnorm_k(const float* __restrict__ x,
                                                 const float* __restrict__ w,
                                                 float* __restrict__ o) {
  __shared__ float red[256];
  const long row = blockIdx.x;
  const float* xr = x + row * D_;
  float s = 0.f;
  for (int i = threadIdx.x; i < D_; i += 256) { const float t = xr[i]; s += t * t; }
  red[threadIdx.x] = s; __syncthreads();
  for (int st = 128; st > 0; st >>= 1) {
    if (threadIdx.x < st) red[threadIdx.x] += red[threadIdx.x + st];
    __syncthreads();
  }
  const float r = rsqrtf(red[0] / (float)D_ + EPS_);
  for (int i = threadIdx.x; i < D_; i += 256) o[row * D_ + i] = xr[i] * r * w[i];
}

__global__ void rope_k(float* __restrict__ x, const float* __restrict__ fr, int H, long n) {
  const long idx = (long)blockIdx.x * 256 + threadIdx.x;
  if (idx >= n) return;
  const int i = idx & 63;                 // HD/2 = 64 pairs
  long tmp = idx >> 6;
  const int h = (int)(tmp % H); tmp /= H;
  const int t = (int)(tmp % T_);
  const int b = (int)(tmp / T_);
  const long base = (((long)(b * T_ + t)) * H + h) * HD_ + 2 * i;
  const float c = fr[(t * 64 + i) * 2 + 0], s = fr[(t * 64 + i) * 2 + 1];
  const float r0 = x[base], r1 = x[base + 1];
  x[base] = r0 * c - r1 * s;
  x[base + 1] = r0 * s + r1 * c;
}

__global__ void vel_k(const float* __restrict__ x, const float* __restrict__ vel,
                      const float* __restrict__ attn, const float* __restrict__ lb,
                      float* __restrict__ xo, float* __restrict__ vo, long n) {
  const long i = (long)blockIdx.x * 256 + threadIdx.x;
  if (i >= n) return;
  const float beta = 1.f / (1.f + __expf(-lb[0]));
  const float vn = beta * vel[i] + attn[i];
  vo[i] = vn;
  xo[i] = x[i] + vn;
}

__global__ void silu_k(const float* __restrict__ gu, float* __restrict__ h, long n) {
  const long i = (long)blockIdx.x * 256 + threadIdx.x;
  if (i >= n) return;
  const long row = i / FI_;
  const int f = (int)(i % FI_);
  const float g = gu[row * (2 * FI_) + f];
  const float u = gu[row * (2 * FI_) + FI_ + f];
  h[i] = g / (1.f + __expf(-g)) * u;
}

__global__ void conv_k(const float* __restrict__ tt, const float* __restrict__ cw,
                       const float* __restrict__ cb, float* __restrict__ o, long n) {
  const long i = (long)blockIdx.x * 256 + threadIdx.x;
  if (i >= n) return;
  const int d = (int)(i % D_);
  const long bt = i / D_;
  const int t = (int)(bt % T_);
  const int b = (int)(bt / T_);
  float acc = cb[d];
#pragma unroll
  for (int j = 0; j < KW_; ++j) {
    const int ti = t - (KW_ - 1) + j;
    if (ti >= 0) acc += tt[((long)(b * T_ + ti)) * D_ + d] * cw[d * KW_ + j];
  }
  o[i] = acc;
}

__global__ void init_wacc(const float* __restrict__ wd, float* __restrict__ wacc, long n) {
  const long i = (long)blockIdx.x * 256 + threadIdx.x;
  if (i >= n) return;
  wacc[i] = wd[i % ((long)D_ * FI_)];
}

// ---------------------------------------------------------------- launch
extern "C" void kernel_launch(void* const* d_in, const int* in_sizes, int n_in,
                              void* d_out, int out_size, void* d_ws, size_t ws_size,
                              hipStream_t stream) {
  const float* x   = (const float*)d_in[0];
  const float* vel = (const float*)d_in[1];
  const float* ttt = (const float*)d_in[2];
  const float* fr  = (const float*)d_in[3];
  const float* pw  = (const float*)d_in[4];
  const float* fw  = (const float*)d_in[5];
  const float* wq  = (const float*)d_in[6];
  const float* wk  = (const float*)d_in[7];
  const float* wv  = (const float*)d_in[8];
  const float* wo  = (const float*)d_in[9];
  const float* lb  = (const float*)d_in[10];
  const float* wgu = (const float*)d_in[11];
  const float* wd  = (const float*)d_in[12];
  const float* tp  = (const float*)d_in[13];
  const float* cw  = (const float*)d_in[14];
  const float* cb  = (const float*)d_in[15];

  const long BTD = (long)B_ * T_ * D_;
  float* out_x = (float*)d_out;
  float* out_v = out_x + BTD;
  float* ws = (float*)d_ws;
  // scratch layout (with lifetime-safe reuse; peak = 79,691,776 floats = ~319 MB)
  float* xn   = ws;                                    // rmsnorm out; reused as ffn_in
  float* qb   = xn + BTD;                              // q; reused as conv output t
  float* kb   = qb + BTD;                              // k
  float* vb   = kb + (long)B_ * T_ * NKV_ * HD_;       // v
  float* yb   = vb + (long)B_ * T_ * NKV_ * HD_;       // attention concat
  float* ab   = yb + BTD;                              // attn_out
  float* gu   = ab + BTD;                              // gate_up
  float* hb   = gu + (long)B_ * T_ * 2 * FI_;          // h = silu(gate)*up
  float* tb   = qb;                                    // conv out (q is dead)
  float* tpj  = kb;                                    // t_proj (k/v/y-head dead)
  float* wacc = ab;                                    // w_acc (attn/gu-head dead)

  const dim3 blk(256);
  long n;

  rmsnorm_k<<<B_ * T_, blk, 0, stream>>>(x, pw, xn);

  gemm_nt<<<dim3((NH_ * HD_) / BN, (B_ * T_) / BM, 1), blk, 0, stream>>>(
      xn, wq, qb, B_ * T_, NH_ * HD_, D_, 0, 0, 0, 1.f, 0);
  gemm_nt<<<dim3((NKV_ * HD_) / BN, (B_ * T_) / BM, 1), blk, 0, stream>>>(
      xn, wk, kb, B_ * T_, NKV_ * HD_, D_, 0, 0, 0, 1.f, 0);
  gemm_nt<<<dim3((NKV_ * HD_) / BN, (B_ * T_) / BM, 1), blk, 0, stream>>>(
      xn, wv, vb, B_ * T_, NKV_ * HD_, D_, 0, 0, 0, 1.f, 0);

  n = (long)B_ * T_ * NH_ * 64;
  rope_k<<<(unsigned)((n + 255) / 256), blk, 0, stream>>>(qb, fr, NH_, n);
  n = (long)B_ * T_ * NKV_ * 64;
  rope_k<<<(unsigned)((n + 255) / 256), blk, 0, stream>>>(kb, fr, NKV_, n);

  attn_fa<<<dim3(T_ / 64, NH_, B_), blk, 0, stream>>>(qb, kb, vb, yb);

  gemm_nt<<<dim3(D_ / BN, (B_ * T_) / BM, 1), blk, 0, stream>>>(
      yb, wo, ab, B_ * T_, D_, NH_ * HD_, 0, 0, 0, 1.f, 0);

  n = BTD;
  vel_k<<<(unsigned)((n + 255) / 256), blk, 0, stream>>>(x, vel, ab, lb, out_x, out_v, n);

  rmsnorm_k<<<B_ * T_, blk, 0, stream>>>(out_x, fw, xn);

  gemm_nt<<<dim3((2 * FI_) / BN, (B_ * T_) / BM, 1), blk, 0, stream>>>(
      xn, wgu, gu, B_ * T_, 2 * FI_, D_, 0, 0, 0, 1.f, 0);

  n = (long)B_ * T_ * FI_;
  silu_k<<<(unsigned)((n + 255) / 256), blk, 0, stream>>>(gu, hb, n);

  n = BTD;
  conv_k<<<(unsigned)((n + 255) / 256), blk, 0, stream>>>(ttt, cw, cb, tb, n);

  // t_proj[b, 0:7*256, :] = TTT_LR * t_chunk . ttt_proj^T  (lr folded -> d_down scaled)
  gemm_nt<<<dim3(D_ / BN, (7 * CHUNK_) / BM, B_), blk, 0, stream>>>(
      tb, tp, tpj, 7 * CHUNK_, D_, D_, (long)T_ * D_, 0, (long)7 * CHUNK_ * D_,
      TTT_LR_, 0);

  n = (long)B_ * D_ * FI_;
  init_wacc<<<(unsigned)((n + 255) / 256), blk, 0, stream>>>(wd, wacc, n);

  for (int i = 0; i < 8; ++i) {
    // out_x[b, chunk i] += h_c . w_acc^T   (cumsum state before this chunk's update)
    gemm_nt<<<dim3(D_ / BN, CHUNK_ / BM, B_), blk, 0, stream>>>(
        hb + (long)i * CHUNK_ * FI_, wacc, out_x + (long)i * CHUNK_ * D_,
        CHUNK_, D_, FI_, (long)T_ * FI_, (long)D_ * FI_, (long)T_ * D_, 1.f, 1);
    if (i < 7) {
      // w_acc += t_proj_chunk^T . h_chunk   (d_down * TTT_LR, lr already folded)
      gemm_tn_acc<<<dim3(FI_ / BN, D_ / BM, B_), blk, 0, stream>>>(
          tpj + (long)i * CHUNK_ * D_, hb + (long)i * CHUNK_ * FI_, wacc,
          D_, FI_, CHUNK_, (long)7 * CHUNK_ * D_, (long)T_ * FI_, (long)D_ * FI_, 1.f);
    }
  }
}
